// Attention_19129784336852
// MI455X (gfx1250) — matrix operand order repaired and checked by reading, not compiled, not run on hardware
//
#include <hip/hip_runtime.h>

// ---------------------------------------------------------------------------
// Fused pre-norm multi-head attention for MI455X (gfx1250, wave32, WMMA).
//   x[8,2048,512] -> LN -> QKV -> flash attention (H=8, Dh=64) -> proj+bias
// All GEMM-like work runs on v_wmma_f32_16x16x32_f16 (f32 accumulate).
// Weight panels are staged into LDS with gfx1250 async-to-LDS copies
// (ASYNCcnt + s_wait_asynccnt) when the toolchain exposes the builtins.
// ---------------------------------------------------------------------------

typedef _Float16 h8   __attribute__((ext_vector_type(8)));
typedef _Float16 v16h __attribute__((ext_vector_type(16)));
typedef float    v8f  __attribute__((ext_vector_type(8)));

// The async-to-LDS builtin takes pointers to GCC-style int4 vectors.
typedef int v4i_vs __attribute__((vector_size(16)));
typedef __attribute__((address_space(1))) v4i_vs* gptr128;
typedef __attribute__((address_space(3))) v4i_vs* lptr128;

#define BATCH  8
#define SEQ    2048
#define DIM    512
#define HEADS  8
#define DHEAD  64
#define INNER  512                 // HEADS * DHEAD
#define TOKENS (BATCH * SEQ)       // 16384
#define QKVN   (3 * INNER)         // 1536
#define WLDSP  520                 // LDS row pitch (halves): 512 + 8 pad, 16B-aligned

#if __has_builtin(__builtin_amdgcn_global_load_async_to_lds_b128) && \
    __has_builtin(__builtin_amdgcn_s_wait_asynccnt)
#define HAVE_ASYNC_LDS 1
#endif

// Two 16-byte loads -> one 16x32 (A) or 32x16 (B) f16 WMMA fragment.
static __device__ __forceinline__ v16h ld2(const _Float16* p0, const _Float16* p1) {
  union { v16h v; h8 h[2]; } u;
  u.h[0] = *(const h8*)p0;
  u.h[1] = *(const h8*)p1;
  return u.v;
}

static __device__ __forceinline__ v8f wmma_f16(v16h a, v16h b, v8f c) {
  // (neg_a, A, neg_b, B, c_mod, C, reuse_a, reuse_b)
  return __builtin_amdgcn_wmma_f32_16x16x32_f16(false, a, false, b, (short)0, c,
                                                false, false);
}

// 16-byte global -> LDS copy; async DMA path when available.
static __device__ __forceinline__ void copy16_g2l(const _Float16* g, _Float16* l) {
#ifdef HAVE_ASYNC_LDS
  __builtin_amdgcn_global_load_async_to_lds_b128(
      (gptr128)(uintptr_t)g, (lptr128)(uintptr_t)l, 0, 0);
#else
  *(h8*)l = *(const h8*)g;
#endif
}

static __device__ __forceinline__ void g2l_wait() {
#ifdef HAVE_ASYNC_LDS
  __builtin_amdgcn_s_wait_asynccnt(0);
#endif
}

// Stage a 64 x 512 f16 weight panel (rows nbase..nbase+63 of a [*,512] f16
// matrix) into LDS with row pitch WLDSP. Cooperative across the block.
static __device__ __forceinline__ void stage_wpanel(const _Float16* wpanel,
                                                    _Float16* lds) {
  // 64 rows x 64 chunks of 8 halves = 4096 chunks, 256 threads -> 16 each.
#pragma unroll
  for (int i = 0; i < 16; ++i) {
    const int c   = i * 256 + threadIdx.x;
    const int row = c >> 6;
    const int cc  = c & 63;
    copy16_g2l(wpanel + (size_t)row * DIM + cc * 8, lds + row * WLDSP + cc * 8);
  }
  g2l_wait();
  __syncthreads();
}

// ---------------------------------------------------------------------------
// Weight f32 -> f16 conversion (grid-stride).
// ---------------------------------------------------------------------------
__global__ void cvt_f16_kernel(const float* __restrict__ in,
                               _Float16* __restrict__ out, int n) {
  int i = blockIdx.x * blockDim.x + threadIdx.x;
  int stride = gridDim.x * blockDim.x;
  for (; i < n; i += stride) out[i] = (_Float16)in[i];
}

// ---------------------------------------------------------------------------
// LayerNorm: one wave per 512-wide row, output f16. 8 rows per block.
// ---------------------------------------------------------------------------
__global__ void __launch_bounds__(256)
ln_kernel(const float* __restrict__ x, const float* __restrict__ gamma,
          const float* __restrict__ beta, _Float16* __restrict__ xn) {
  const int lane = threadIdx.x & 31;
  const int wave = threadIdx.x >> 5;
  const int row  = blockIdx.x * 8 + wave;
  const float* xr = x + (size_t)row * DIM;
  const int c0 = lane * 16;

  float v[16];
  float s = 0.f, sq = 0.f;
#pragma unroll
  for (int i = 0; i < 4; ++i) {
    float4 f = *(const float4*)(xr + c0 + i * 4);
    v[i * 4 + 0] = f.x; v[i * 4 + 1] = f.y; v[i * 4 + 2] = f.z; v[i * 4 + 3] = f.w;
    s  += f.x + f.y + f.z + f.w;
    sq += f.x * f.x + f.y * f.y + f.z * f.z + f.w * f.w;
  }
#pragma unroll
  for (int m = 16; m >= 1; m >>= 1) {
    s  += __shfl_xor(s,  m, 32);
    sq += __shfl_xor(sq, m, 32);
  }
  const float mu  = s * (1.0f / DIM);
  const float var = sq * (1.0f / DIM) - mu * mu;
  const float rs  = rsqrtf(var + 1e-6f);

  h8 o0, o1;
#pragma unroll
  for (int i = 0; i < 16; ++i) {
    float t = (v[i] - mu) * rs * gamma[c0 + i] + beta[c0 + i];
    if (i < 8) o0[i] = (_Float16)t; else o1[i - 8] = (_Float16)t;
  }
  _Float16* orow = xn + (size_t)row * DIM + c0;
  *(h8*)(orow)     = o0;
  *(h8*)(orow + 8) = o1;
}

// ---------------------------------------------------------------------------
// QKV GEMM: qkv = xn @ w_qkv^T. Each wave: 16 rows x 64 cols, K=512 in 32-steps.
// The 64x512 weight panel is staged in LDS (async DMA), shared by all 8 waves.
// Q,K stored [b][h][n][64] f16; V stored transposed [b][h][64][n] f16 so the
// PV WMMA B-fragment is a contiguous load later.
// ---------------------------------------------------------------------------
__global__ void __launch_bounds__(256)
qkv_kernel(const _Float16* __restrict__ xn, const _Float16* __restrict__ w,
           _Float16* __restrict__ Qb, _Float16* __restrict__ Kb,
           _Float16* __restrict__ Vt) {
  __shared__ __align__(16) _Float16 wlds[64 * WLDSP];      // 65 KB panel

  const int lane  = threadIdx.x & 31;
  const int wave  = threadIdx.x >> 5;
  const int nbase = blockIdx.x * 64;                       // 0..1472
  const int m0    = (blockIdx.y * 8 + wave) * 16;          // 16-row strip
  const int lrow  = lane & 15;
  const int hiA   = (lane & 16) ? 8 : 0;                   // A-frag K sub-offset
  const int hiB   = (lane & 16) ? 8 : 0;                   // B-frag K sub-offset
  const int rowoff = (lane & 16) ? 8 : 0;                  // C-frag row offset

  stage_wpanel(w + (size_t)nbase * DIM, wlds);

  v8f acc[4] = {};
  const _Float16* arow = xn + (size_t)(m0 + lrow) * DIM;
  const _Float16* bbase = wlds + lrow * WLDSP;

  for (int kk = 0; kk < DIM; kk += 32) {
    __builtin_prefetch(arow + kk + 64, 0, 0);              // global_prefetch_b8
    v16h a = ld2(arow + kk + hiA, arow + kk + hiA + 16);
#pragma unroll
    for (int t = 0; t < 4; ++t) {
      const _Float16* brow = bbase + t * (16 * WLDSP) + kk + hiB;
      v16h bf = ld2(brow, brow + 16);                      // ds_load_b128 x2
      acc[t] = wmma_f16(a, bf, acc[t]);
    }
  }

#pragma unroll
  for (int t = 0; t < 4; ++t) {
    const int col  = nbase + t * 16 + lrow;                // 0..1535
    const int part = col >> 9;                             // 0=Q 1=K 2=V
    const int cc   = col & 511;
    const int h    = cc >> 6;
    const int d    = cc & 63;
#pragma unroll
    for (int r = 0; r < 8; ++r) {
      const int m = m0 + r + rowoff;
      const int b = m >> 11;
      const int n = m & (SEQ - 1);
      const size_t bh = (size_t)(b * HEADS + h);
      const _Float16 val = (_Float16)acc[t][r];
      if (part == 0)      Qb[(bh * SEQ + n) * DHEAD + d] = val;
      else if (part == 1) Kb[(bh * SEQ + n) * DHEAD + d] = val;
      else                Vt[(bh * DHEAD + d) * SEQ + n] = val;
    }
  }
}

// ---------------------------------------------------------------------------
// Flash attention. Block = 8 waves; each wave owns a 16-query strip of one
// (b,h) and streams all 2048 keys in 32-key blocks:
//   S (2 tiles, 4 WMMA) -> scaled online softmax (shfl over 16-lane C groups)
//   -> P relayout via per-wave LDS (C-frag -> A-frag) -> PV (4 WMMA).
// Output written f16 [b][n][h*64+d] for the projection GEMM.
// ---------------------------------------------------------------------------
__global__ void __launch_bounds__(256)
attn_kernel(const _Float16* __restrict__ Qb, const _Float16* __restrict__ Kb,
            const _Float16* __restrict__ Vt, _Float16* __restrict__ Ob) {
  __shared__ __align__(16) _Float16 lds[8 * 16 * 40];      // per-wave P buffer

  const int lane = threadIdx.x & 31;
  const int wave = threadIdx.x >> 5;
  const int bh   = blockIdx.y;                             // 0..63
  const int b    = bh >> 3;
  const int h    = bh & 7;
  const int q0   = blockIdx.x * 128 + wave * 16;
  const int lrow = lane & 15;
  const int hiA  = (lane & 16) ? 8 : 0;
  const int hiB  = (lane & 16) ? 8 : 0;
  const int rowoff = (lane & 16) ? 8 : 0;
  const float scale = 0.125f;                              // 64^-0.5

  // Q strip -> two A fragments (d = 0..31, 32..63)
  const _Float16* qrow = Qb + ((size_t)bh * SEQ + q0 + lrow) * DHEAD;
  const v16h aq0 = ld2(qrow + hiA,      qrow + hiA + 16);
  const v16h aq1 = ld2(qrow + 32 + hiA, qrow + 32 + hiA + 16);

  v8f o[4] = {};
  float mi[8], li[8];
#pragma unroll
  for (int r = 0; r < 8; ++r) { mi[r] = -1e30f; li[r] = 0.f; }

  _Float16* pbuf = lds + wave * (16 * 40);
  const _Float16* kbase = Kb + (size_t)bh * SEQ * DHEAD;
  const _Float16* vbase = Vt + (size_t)bh * DHEAD * SEQ;

  for (int kb = 0; kb < SEQ; kb += 32) {
    // ---- S = Q K^T for 32 keys (two 16x16 tiles) ----
    v8f s[2];
#pragma unroll
    for (int t = 0; t < 2; ++t) {
      const _Float16* krow = kbase + (size_t)(kb + t * 16 + lrow) * DHEAD;
      v16h bk0 = ld2(krow + hiB,      krow + hiB + 16);
      v16h bk1 = ld2(krow + 32 + hiB, krow + 32 + hiB + 16);
      v8f z = {};
      z = wmma_f16(aq0, bk0, z);
      z = wmma_f16(aq1, bk1, z);
      s[t] = z;
    }

    // ---- online softmax: block row-max, rescale factor ----
    float mnew[8], alpha[8];
#pragma unroll
    for (int r = 0; r < 8; ++r) {
      const float x0 = s[0][r] * scale;
      const float x1 = s[1][r] * scale;
      s[0][r] = x0; s[1][r] = x1;
      float mx = fmaxf(x0, x1);
      mx = fmaxf(mx, __shfl_xor(mx, 1, 32));
      mx = fmaxf(mx, __shfl_xor(mx, 2, 32));
      mx = fmaxf(mx, __shfl_xor(mx, 4, 32));
      mx = fmaxf(mx, __shfl_xor(mx, 8, 32));
      const float mn = fmaxf(mi[r], mx);
      alpha[r] = __expf(mi[r] - mn);
      mnew[r]  = mn;
      mi[r]    = mn;
    }

    // ---- P = exp(S - m), row-sum, l update, spill P (f16) to LDS ----
#pragma unroll
    for (int r = 0; r < 8; ++r) {
      const float p0 = __expf(s[0][r] - mnew[r]);
      const float p1 = __expf(s[1][r] - mnew[r]);
      float sm = p0 + p1;
      sm += __shfl_xor(sm, 1, 32);
      sm += __shfl_xor(sm, 2, 32);
      sm += __shfl_xor(sm, 4, 32);
      sm += __shfl_xor(sm, 8, 32);
      li[r] = li[r] * alpha[r] + sm;
      const int prow = r + rowoff;                          // C-frag row
      pbuf[prow * 40 + lrow]      = (_Float16)p0;           // keys kb..kb+15
      pbuf[prow * 40 + 16 + lrow] = (_Float16)p1;           // keys kb+16..kb+31
    }

    // ---- rescale running output ----
#pragma unroll
    for (int t = 0; t < 4; ++t)
#pragma unroll
      for (int r = 0; r < 8; ++r) o[t][r] *= alpha[r];

    // ---- read P back as a 16x32 A fragment (DS in-order within wave) ----
    const _Float16* prd = pbuf + lrow * 40 + hiA;
    const v16h pa = ld2(prd, prd + 16);

    // ---- O += P V : B-fragments are contiguous rows of V^T ----
#pragma unroll
    for (int t = 0; t < 4; ++t) {
      const _Float16* vrow = vbase + (size_t)(t * 16 + lrow) * SEQ + kb + hiB;
      v16h bv = ld2(vrow, vrow + 16);
      o[t] = wmma_f16(pa, bv, o[t]);
    }
  }

  // ---- normalize and store f16 [b][n][h*64+d] ----
  float inv[8];
#pragma unroll
  for (int r = 0; r < 8; ++r) inv[r] = 1.0f / li[r];
#pragma unroll
  for (int t = 0; t < 4; ++t)
#pragma unroll
    for (int r = 0; r < 8; ++r) {
      const int tok = q0 + r + rowoff;
      Ob[((size_t)(b * SEQ + tok)) * INNER + h * DHEAD + t * 16 + lrow] =
          (_Float16)(o[t][r] * inv[r]);
    }
}

// ---------------------------------------------------------------------------
// Output projection: out = attnO @ w_proj^T + b_proj (f32 output).
// Weight panel staged in LDS (async DMA); bias folded into accumulator init.
// ---------------------------------------------------------------------------
__global__ void __launch_bounds__(256)
proj_kernel(const _Float16* __restrict__ A, const _Float16* __restrict__ W,
            const float* __restrict__ bias, float* __restrict__ out) {
  __shared__ __align__(16) _Float16 wlds[64 * WLDSP];

  const int lane  = threadIdx.x & 31;
  const int wave  = threadIdx.x >> 5;
  const int nbase = blockIdx.x * 64;                       // 0..448
  const int m0    = (blockIdx.y * 8 + wave) * 16;
  const int lrow  = lane & 15;
  const int hiA   = (lane & 16) ? 8 : 0;
  const int hiB   = (lane & 16) ? 8 : 0;
  const int rowoff = (lane & 16) ? 8 : 0;

  stage_wpanel(W + (size_t)nbase * INNER, wlds);

  v8f acc[4];
#pragma unroll
  for (int t = 0; t < 4; ++t) {
    const float bv = bias[nbase + t * 16 + lrow];
#pragma unroll
    for (int r = 0; r < 8; ++r) acc[t][r] = bv;
  }

  const _Float16* arow = A + (size_t)(m0 + lrow) * INNER;
  const _Float16* bbase = wlds + lrow * WLDSP;

  for (int kk = 0; kk < INNER; kk += 32) {
    __builtin_prefetch(arow + kk + 64, 0, 0);
    v16h a = ld2(arow + kk + hiA, arow + kk + hiA + 16);
#pragma unroll
    for (int t = 0; t < 4; ++t) {
      const _Float16* brow = bbase + t * (16 * WLDSP) + kk + hiB;
      v16h bf = ld2(brow, brow + 16);
      acc[t] = wmma_f16(a, bf, acc[t]);
    }
  }

#pragma unroll
  for (int t = 0; t < 4; ++t)
#pragma unroll
    for (int r = 0; r < 8; ++r)
      out[(size_t)(m0 + r + rowoff) * DIM + nbase + t * 16 + lrow] = acc[t][r];
}

// ---------------------------------------------------------------------------
extern "C" void kernel_launch(void* const* d_in, const int* in_sizes, int n_in,
                              void* d_out, int out_size, void* d_ws, size_t ws_size,
                              hipStream_t stream) {
  (void)in_sizes; (void)n_in; (void)out_size; (void)ws_size;
  const float* x      = (const float*)d_in[0];
  const float* w_qkv  = (const float*)d_in[1];
  const float* w_proj = (const float*)d_in[2];
  const float* b_proj = (const float*)d_in[3];
  const float* gamma  = (const float*)d_in[4];
  const float* beta   = (const float*)d_in[5];
  float* out = (float*)d_out;

  char* ws = (char*)d_ws;
  size_t off = 0;
  auto alloc = [&](size_t bytes) -> void* {
    void* p = ws + off;
    off += (bytes + 255) & ~(size_t)255;
    return p;
  };
  _Float16* xn   = (_Float16*)alloc((size_t)TOKENS * DIM * 2);     // 16 MB
  _Float16* wq16 = (_Float16*)alloc((size_t)QKVN * DIM * 2);       // 1.5 MB
  _Float16* wp16 = (_Float16*)alloc((size_t)DIM * INNER * 2);      // 0.5 MB
  _Float16* Qb   = (_Float16*)alloc((size_t)BATCH * HEADS * SEQ * DHEAD * 2);
  _Float16* Kb   = (_Float16*)alloc((size_t)BATCH * HEADS * SEQ * DHEAD * 2);
  _Float16* Vt   = (_Float16*)alloc((size_t)BATCH * HEADS * DHEAD * SEQ * 2);
  _Float16* Ob   = (_Float16*)alloc((size_t)TOKENS * INNER * 2);   // 16 MB

  cvt_f16_kernel<<<dim3(512), dim3(256), 0, stream>>>(w_qkv, wq16, QKVN * DIM);
  cvt_f16_kernel<<<dim3(256), dim3(256), 0, stream>>>(w_proj, wp16, DIM * INNER);
  ln_kernel<<<dim3(TOKENS / 8), dim3(256), 0, stream>>>(x, gamma, beta, xn);
  qkv_kernel<<<dim3(QKVN / 64, TOKENS / 128), dim3(256), 0, stream>>>(xn, wq16, Qb, Kb, Vt);
  attn_kernel<<<dim3(SEQ / 128, BATCH * HEADS), dim3(256), 0, stream>>>(Qb, Kb, Vt, Ob);
  proj_kernel<<<dim3(DIM / 64, TOKENS / 128), dim3(256), 0, stream>>>(Ob, wp16, b_proj, out);
}
